// AttentionBlock_vec_13314398618116
// MI455X (gfx1250) — compile-verified, matrix operand order checked by reading, not run on hardware
//
#include <hip/hip_runtime.h>
#include <math.h>
#include <stdint.h>

typedef __attribute__((ext_vector_type(16))) _Float16 v16h;
typedef __attribute__((ext_vector_type(8)))  float    v8f;

struct alignas(16) H8 { _Float16 h[8]; };

constexpr int Bb = 8, T = 1024, Dd = 1024, Hh = 16, DH = 64;
constexpr int Mrows = Bb * T;          // 8192
constexpr float EPS = 1e-5f;

// ---------------------------------------------------------------- convert f32 -> f16 with transpose
// in:  W  [K=1024][N=1024] f32   out: Wt [N=1024][K=1024] f16   (Wt[n][k] = W[k][n])
__global__ void __launch_bounds__(256)
cvt_t_kernel(const float* __restrict__ W, _Float16* __restrict__ Wt)
{
    constexpr int N = Dd, K = Dd, TS = 72;       // padded LDS stride (144B, 16B aligned)
    __shared__ _Float16 tile[64 * TS];           // [n_local][k_local]
    const int tid = threadIdx.x;
    const int bn = blockIdx.x * 64;              // n tile origin
    const int bk = blockIdx.y * 64;              // k tile origin
#pragma unroll
    for (int t = 0; t < 4; ++t) {                // 64x64 f32 = 1024 float4, 4/thread
        int i = tid + t * 256;
        int r = i >> 4, c4 = (i & 15) * 4;       // r = k_local, c4 = n_local
        float4 f = *reinterpret_cast<const float4*>(W + (size_t)(bk + r) * N + bn + c4);
        tile[(c4 + 0) * TS + r] = (_Float16)f.x;
        tile[(c4 + 1) * TS + r] = (_Float16)f.y;
        tile[(c4 + 2) * TS + r] = (_Float16)f.z;
        tile[(c4 + 3) * TS + r] = (_Float16)f.w;
    }
    __syncthreads();
#pragma unroll
    for (int t = 0; t < 2; ++t) {                // 64x64 halves = 512 h8, 2/thread
        int i = tid + t * 256;
        int r = i >> 3, c8 = (i & 7) * 8;        // r = n_local, c8 = k_local
        H8 o = *reinterpret_cast<const H8*>(&tile[r * TS + c8]);
        *reinterpret_cast<H8*>(Wt + (size_t)(bn + r) * K + bk + c8) = o;
    }
}

// ---------------------------------------------------------------- layernorm (opt. residual add)
template<bool ADD>
__global__ void __launch_bounds__(256)
ln_kernel(const float* __restrict__ x, const float* __restrict__ yres,
          const float* __restrict__ g, const float* __restrict__ be,
          float* __restrict__ xr, _Float16* __restrict__ out)
{
    const int row = blockIdx.x, tid = threadIdx.x;
    const size_t base = (size_t)row * Dd;
    float v[4];
#pragma unroll
    for (int i = 0; i < 4; ++i) {
        int d = tid + i * 256;
        float t = x[base + d];
        if (ADD) { t += yres[base + d]; xr[base + d] = t; }
        v[i] = t;
    }
    float s  = v[0] + v[1] + v[2] + v[3];
    float sq = v[0]*v[0] + v[1]*v[1] + v[2]*v[2] + v[3]*v[3];
#pragma unroll
    for (int m = 1; m < 32; m <<= 1) {
        s  += __shfl_xor(s,  m, 32);
        sq += __shfl_xor(sq, m, 32);
    }
    __shared__ float ss[8], ssq[8];
    int wid = tid >> 5, lane = tid & 31;
    if (lane == 0) { ss[wid] = s; ssq[wid] = sq; }
    __syncthreads();
    s = 0.f; sq = 0.f;
#pragma unroll
    for (int w = 0; w < 8; ++w) { s += ss[w]; sq += ssq[w]; }
    float mean = s * (1.f / Dd);
    float var  = sq * (1.f / Dd) - mean * mean;
    float rstd = rsqrtf(var + EPS);
#pragma unroll
    for (int i = 0; i < 4; ++i) {
        int d = tid + i * 256;
        out[base + d] = (_Float16)((v[i] - mean) * rstd * g[d] + be[d]);
    }
}

// ---------------------------------------------------------------- WMMA GEMM with async-to-LDS double buffering
// A: [8192,1024] f16 row-major ; Wt: [N=1024][K=1024] f16 row-major (pre-transposed weight)
// C = act((A@W + bias)*scale) [+resid] ; block tile 128x64, 8 waves, BK=32.
template<int ACT, int OUTF32>
__global__ void __launch_bounds__(256)
gemm_kernel(const _Float16* __restrict__ A, const _Float16* __restrict__ Wt,
            const float* __restrict__ bias, float scale,
            const float* __restrict__ resid,
            _Float16* __restrict__ outh, float* __restrict__ outf)
{
    constexpr int N = Dd, K = Dd, BM = 128, BN = 64, BK = 32, AS = 40; // padded LDS stride
    __shared__ _Float16 As [2][BM * AS];
    __shared__ _Float16 Bst[2][BN * AS];   // [col n][k]
    const int tid  = threadIdx.x;
    const int wid  = tid >> 5, lane = tid & 31, hf = lane >> 4, ln = lane & 15;
    const int brow = blockIdx.x * BM, bcol = blockIdx.y * BN;
    const int koff = hf * 8;

    // issue one K-tile's async global->LDS copies (3 x b128 per thread)
    auto issue_tile = [&](int buf, int kb) {
#pragma unroll
        for (int t = 0; t < 2; ++t) {           // A tile: 128x32 halves = 512 b128 chunks
            int i = tid + t * 256;
            int r = i >> 2, c = (i & 3) * 8;
            unsigned voff = (unsigned)(((brow + r) * K + kb + c) * 2);
            unsigned lds  = (unsigned)(uintptr_t)&As[buf][r * AS + c];
            asm volatile("global_load_async_to_lds_b128 %0, %1, %2"
                         :: "v"(lds), "v"(voff), "s"(A) : "memory");
        }
        {                                       // B tile: 64x32 halves = 256 b128 chunks
            int r = tid >> 2, c = (tid & 3) * 8;
            unsigned voff = (unsigned)(((bcol + r) * K + kb + c) * 2);
            unsigned lds  = (unsigned)(uintptr_t)&Bst[buf][r * AS + c];
            asm volatile("global_load_async_to_lds_b128 %0, %1, %2"
                         :: "v"(lds), "v"(voff), "s"(Wt) : "memory");
        }
    };

    v8f acc[4] = {};

    issue_tile(0, 0);
    asm volatile("s_wait_asynccnt 0x0" ::: "memory");
    __syncthreads();

    for (int kb = 0; kb < K; kb += BK) {
        const int cur = (kb >> 5) & 1;
        if (kb + BK < K) issue_tile(cur ^ 1, kb + BK);   // overlap next tile with compute

        // A fragment (shared across the wave's 4 output tiles)
        v16h af;
        {
            H8 lo = *reinterpret_cast<const H8*>(&As[cur][(wid * 16 + ln) * AS + koff]);
            H8 hi = *reinterpret_cast<const H8*>(&As[cur][(wid * 16 + ln) * AS + 16 + koff]);
#pragma unroll
            for (int i = 0; i < 8; ++i) { af[i] = lo.h[i]; af[8 + i] = hi.h[i]; }
        }
#pragma unroll
        for (int j = 0; j < 4; ++j) {
            H8 blo = *reinterpret_cast<const H8*>(&Bst[cur][(j * 16 + ln) * AS + koff]);
            H8 bhi = *reinterpret_cast<const H8*>(&Bst[cur][(j * 16 + ln) * AS + 16 + koff]);
            v16h bf;
#pragma unroll
            for (int i = 0; i < 8; ++i) { bf[i] = blo.h[i]; bf[8 + i] = bhi.h[i]; }
            acc[j] = __builtin_amdgcn_wmma_f32_16x16x32_f16(false, af, false, bf,
                                                            (short)0, acc[j], false, false);
        }

        asm volatile("s_wait_asynccnt 0x0" ::: "memory");  // own async copies landed
        __syncthreads();                                    // everyone's copies + compute done
    }

#pragma unroll
    for (int j = 0; j < 4; ++j) {
        int col = bcol + j * 16 + ln;
        float bv = bias[col];
#pragma unroll
        for (int r = 0; r < 8; ++r) {
            int row = brow + wid * 16 + hf * 8 + r;
            float v = (acc[j][r] + bv) * scale;
            if (ACT) v = fmaxf(v, 0.f);
            if (OUTF32) outf[(size_t)row * N + col] = v + resid[(size_t)row * N + col];
            else        outh[(size_t)row * N + col] = (_Float16)v;
        }
    }
}

// ---------------------------------------------------------------- flash attention: y = softmax(K@Q^T) @ V
// Note the reference role swap: K rows index the output. Grid (T/64, B*H), 128 threads (4 waves).
__global__ void __launch_bounds__(128)
attn_kernel(const _Float16* __restrict__ qf, const _Float16* __restrict__ kf,
            const _Float16* __restrict__ vf, float* __restrict__ y)
{
    constexpr int LS = 72;   // padded stride (144B, 16B-aligned h8 reads)
    __shared__ _Float16 Ks [64 * LS];
    __shared__ _Float16 Qs [64 * LS];
    __shared__ _Float16 Vst[64 * LS];       // transposed: [d][s]
    __shared__ _Float16 Ps [4][16 * LS];    // wave-private P tiles
    const int tid = threadIdx.x;
    const int wid = tid >> 5, lane = tid & 31, hf = lane >> 4, ln = lane & 15;
    const int t0 = blockIdx.x * 64;
    const int b  = blockIdx.y >> 4, h = blockIdx.y & 15;
    const size_t rowbase = (size_t)b * T;
    const int dcol0 = h * DH;
    const int koff = hf * 8;

    // stage this block's 64 K-rows once (already pre-scaled by 1/sqrt(DH))
    for (int i = tid; i < 64 * 64 / 8; i += 128) {
        int r = i >> 3, c8 = (i & 7) * 8;
        uint4 vK = *reinterpret_cast<const uint4*>(kf + (rowbase + t0 + r) * Dd + dcol0 + c8);
        *reinterpret_cast<uint4*>(&Ks[r * LS + c8]) = vK;
    }
    __syncthreads();

    v16h kA[2];   // K fragments are loop-invariant: preload both dh k-steps
#pragma unroll
    for (int ks = 0; ks < 2; ++ks) {
        H8 lo = *reinterpret_cast<const H8*>(&Ks[(wid * 16 + ln) * LS + ks * 32 + koff]);
        H8 hi = *reinterpret_cast<const H8*>(&Ks[(wid * 16 + ln) * LS + ks * 32 + 16 + koff]);
#pragma unroll
        for (int i = 0; i < 8; ++i) { kA[ks][i] = lo.h[i]; kA[ks][8 + i] = hi.h[i]; }
    }

    v8f Yacc[4] = {};
    float mrow[8], lrow[8];
#pragma unroll
    for (int r = 0; r < 8; ++r) { mrow[r] = -INFINITY; lrow[r] = 0.f; }

    for (int s0 = 0; s0 < T; s0 += 64) {
        __syncthreads();
        for (int i = tid; i < 64 * 64 / 8; i += 128) {
            int r = i >> 3, c8 = (i & 7) * 8;
            uint4 vQ = *reinterpret_cast<const uint4*>(qf + (rowbase + s0 + r) * Dd + dcol0 + c8);
            *reinterpret_cast<uint4*>(&Qs[r * LS + c8]) = vQ;
            H8 w = *reinterpret_cast<const H8*>(vf + (rowbase + s0 + r) * Dd + dcol0 + c8);
#pragma unroll
            for (int j = 0; j < 8; ++j) Vst[(c8 + j) * LS + r] = w.h[j];
        }
        __syncthreads();

        // S tiles: A = K rows, B = Q^T (Qs natural [s][d] layout gives contiguous B frags)
        v8f Sc[4] = {};
#pragma unroll
        for (int j = 0; j < 4; ++j) {
#pragma unroll
            for (int ks = 0; ks < 2; ++ks) {
                H8 blo = *reinterpret_cast<const H8*>(&Qs[(j * 16 + ln) * LS + ks * 32 + koff]);
                H8 bhi = *reinterpret_cast<const H8*>(&Qs[(j * 16 + ln) * LS + ks * 32 + 16 + koff]);
                v16h bf;
#pragma unroll
                for (int i = 0; i < 8; ++i) { bf[i] = blo.h[i]; bf[8 + i] = bhi.h[i]; }
                Sc[j] = __builtin_amdgcn_wmma_f32_16x16x32_f16(false, kA[ks], false, bf,
                                                               (short)0, Sc[j], false, false);
            }
        }

        // online softmax: rows live across 16-lane halves -> butterfly masks 1/2/4/8
#pragma unroll
        for (int r = 0; r < 8; ++r) {
            float rm = fmaxf(fmaxf(Sc[0][r], Sc[1][r]), fmaxf(Sc[2][r], Sc[3][r]));
            rm = fmaxf(rm, __shfl_xor(rm, 1, 32));
            rm = fmaxf(rm, __shfl_xor(rm, 2, 32));
            rm = fmaxf(rm, __shfl_xor(rm, 4, 32));
            rm = fmaxf(rm, __shfl_xor(rm, 8, 32));
            float mn = fmaxf(mrow[r], rm);
            float sf = __expf(mrow[r] - mn);
            mrow[r] = mn;
            lrow[r] *= sf;
#pragma unroll
            for (int j = 0; j < 4; ++j) Yacc[j][r] *= sf;
            float racc = 0.f;
#pragma unroll
            for (int j = 0; j < 4; ++j) {
                float p = __expf(Sc[j][r] - mn);
                Ps[wid][(hf * 8 + r) * LS + j * 16 + ln] = (_Float16)p;  // C-layout -> LDS
                racc += p;
            }
            racc += __shfl_xor(racc, 1, 32);
            racc += __shfl_xor(racc, 2, 32);
            racc += __shfl_xor(racc, 4, 32);
            racc += __shfl_xor(racc, 8, 32);
            lrow[r] += racc;
        }

        // Y += P @ V  (A frags re-read from wave-private LDS in A-layout; B from transposed V)
#pragma unroll
        for (int ks = 0; ks < 2; ++ks) {
            v16h pf;
            H8 plo = *reinterpret_cast<const H8*>(&Ps[wid][ln * LS + ks * 32 + koff]);
            H8 phi = *reinterpret_cast<const H8*>(&Ps[wid][ln * LS + ks * 32 + 16 + koff]);
#pragma unroll
            for (int i = 0; i < 8; ++i) { pf[i] = plo.h[i]; pf[8 + i] = phi.h[i]; }
#pragma unroll
            for (int j = 0; j < 4; ++j) {
                H8 blo = *reinterpret_cast<const H8*>(&Vst[(j * 16 + ln) * LS + ks * 32 + koff]);
                H8 bhi = *reinterpret_cast<const H8*>(&Vst[(j * 16 + ln) * LS + ks * 32 + 16 + koff]);
                v16h bf;
#pragma unroll
                for (int i = 0; i < 8; ++i) { bf[i] = blo.h[i]; bf[8 + i] = bhi.h[i]; }
                Yacc[j] = __builtin_amdgcn_wmma_f32_16x16x32_f16(false, pf, false, bf,
                                                                 (short)0, Yacc[j], false, false);
            }
        }
    }

#pragma unroll
    for (int j = 0; j < 4; ++j) {
        int col = dcol0 + j * 16 + ln;
#pragma unroll
        for (int r = 0; r < 8; ++r) {
            int t = t0 + wid * 16 + hf * 8 + r;
            y[(rowbase + t) * Dd + col] = Yacc[j][r] / lrow[r];
        }
    }
}

// ---------------------------------------------------------------- host orchestration
extern "C" void kernel_launch(void* const* d_in, const int* in_sizes, int n_in,
                              void* d_out, int out_size, void* d_ws, size_t ws_size,
                              hipStream_t stream)
{
    const float* x   = (const float*)d_in[0];
    const float* Wq  = (const float*)d_in[1];  const float* bq = (const float*)d_in[2];
    const float* Wk  = (const float*)d_in[3];  const float* bk = (const float*)d_in[4];
    const float* Wv  = (const float*)d_in[5];  const float* bv = (const float*)d_in[6];
    const float* g1  = (const float*)d_in[7];  const float* be1 = (const float*)d_in[8];
    const float* g2  = (const float*)d_in[9];  const float* be2 = (const float*)d_in[10];
    const float* W0  = (const float*)d_in[11]; const float* b0 = (const float*)d_in[12];
    const float* W1  = (const float*)d_in[13]; const float* b1 = (const float*)d_in[14];
    const float* W2  = (const float*)d_in[15]; const float* b2 = (const float*)d_in[16];
    float* out = (float*)d_out;

    char* ws = (char*)d_ws;
    size_t off = 0;
    auto take = [&](size_t bytes) -> char* {
        char* p = ws + off;
        off = (off + bytes + 255) & ~(size_t)255;
        return p;
    };
    const size_t WH = (size_t)Dd * Dd * sizeof(_Float16);
    const size_t MH = (size_t)Mrows * Dd * sizeof(_Float16);
    const size_t MF = (size_t)Mrows * Dd * sizeof(float);

    _Float16* Wqh = (_Float16*)take(WH);   // all weights stored TRANSPOSED [N][K] f16
    _Float16* Wkh = (_Float16*)take(WH);
    _Float16* Wvh = (_Float16*)take(WH);
    _Float16* W0h = (_Float16*)take(WH);
    _Float16* W1h = (_Float16*)take(WH);
    _Float16* W2h = (_Float16*)take(WH);
    _Float16* xn1 = (_Float16*)take(MH);
    _Float16* qh  = (_Float16*)take(MH);
    _Float16* kh  = (_Float16*)take(MH);
    _Float16* vh  = (_Float16*)take(MH);
    float*    y32 = (float*)take(MF);
    float*    xr  = (float*)take(MF);
    _Float16* xn2 = xn1;  // reuse after QKV consumed
    _Float16* h1  = qh;   // reuse after attention consumed
    _Float16* h2  = kh;

    dim3 cg(Dd / 64, Dd / 64);
    cvt_t_kernel<<<cg, 256, 0, stream>>>(Wq, Wqh);
    cvt_t_kernel<<<cg, 256, 0, stream>>>(Wk, Wkh);
    cvt_t_kernel<<<cg, 256, 0, stream>>>(Wv, Wvh);
    cvt_t_kernel<<<cg, 256, 0, stream>>>(W0, W0h);
    cvt_t_kernel<<<cg, 256, 0, stream>>>(W1, W1h);
    cvt_t_kernel<<<cg, 256, 0, stream>>>(W2, W2h);

    ln_kernel<false><<<Mrows, 256, 0, stream>>>(x, x, g1, be1, xr, xn1);

    dim3 gg(Mrows / 128, Dd / 64);
    gemm_kernel<0, 0><<<gg, 256, 0, stream>>>(xn1, Wqh, bq, 1.f,    xr, qh, y32);
    gemm_kernel<0, 0><<<gg, 256, 0, stream>>>(xn1, Wkh, bk, 0.125f, xr, kh, y32); // 1/sqrt(DH)
    gemm_kernel<0, 0><<<gg, 256, 0, stream>>>(xn1, Wvh, bv, 1.f,    xr, vh, y32);

    attn_kernel<<<dim3(T / 64, Bb * Hh), 128, 0, stream>>>(qh, kh, vh, y32);

    ln_kernel<true><<<Mrows, 256, 0, stream>>>(x, y32, g2, be2, xr, xn2);

    gemm_kernel<1, 0><<<gg, 256, 0, stream>>>(xn2, W0h, b0, 1.f, xr, h1, out);
    gemm_kernel<1, 0><<<gg, 256, 0, stream>>>(h1,  W1h, b1, 1.f, xr, h2, out);
    gemm_kernel<0, 1><<<gg, 256, 0, stream>>>(h2,  W2h, b2, 1.f, xr, h2, out);
}
